// MultiHeadAttention_13812614824392
// MI455X (gfx1250) — matrix operand order repaired and checked by reading, not compiled, not run on hardware
//
#include <hip/hip_runtime.h>

typedef _Float16 half_t;
typedef __attribute__((ext_vector_type(16))) _Float16 v16h;
typedef __attribute__((ext_vector_type(8)))  _Float16 v8h;
typedef __attribute__((ext_vector_type(4)))  _Float16 v4h;
typedef __attribute__((ext_vector_type(8)))  float    v8f;
typedef __attribute__((ext_vector_type(4)))  float    v4f;

#define D_MODEL 1024
#define NHEAD   16
#define DK      64
#define SEQ     2048
#define BATCH   2
#define NTOK    (BATCH*SEQ)   // 4096
#define BAND    100

static __device__ __forceinline__ v8f wmma16x16x32(v16h a, v16h b, v8f c) {
  // D(16x16,f32) = A(16x32,f16) * B(32x16,f16) + C
  return __builtin_amdgcn_wmma_f32_16x16x32_f16(false, a, false, b, (short)0, c,
                                                false, false);
}

union H16 { v16h v; v8h h[2]; };

// ---- DPP16 butterfly reduction within each 16-lane half of the wave32 ----
template<int CTRL>
static __device__ __forceinline__ float dpp_permf(float x) {
  int i = __float_as_int(x);
  return __int_as_float(__builtin_amdgcn_update_dpp(i, i, CTRL, 0xF, 0xF, true));
}
static __device__ __forceinline__ float rowmax16(float v) {
  v = fmaxf(v, dpp_permf<0xB1>(v));    // quad_perm(1,0,3,2)
  v = fmaxf(v, dpp_permf<0x4E>(v));    // quad_perm(2,3,0,1)
  v = fmaxf(v, dpp_permf<0x124>(v));   // row_ror:4
  v = fmaxf(v, dpp_permf<0x128>(v));   // row_ror:8
  return v;
}
static __device__ __forceinline__ float rowsum16(float v) {
  v += dpp_permf<0xB1>(v);
  v += dpp_permf<0x4E>(v);
  v += dpp_permf<0x124>(v);
  v += dpp_permf<0x128>(v);
  return v;
}

// A fragment 16x32 (MxK) from row-major f16 [16 rows x stride].
// lane m(0-15): row m, K=0..7 & 16..23 ; lane m+16: row m, K=8..15 & 24..31
static __device__ __forceinline__ v16h load_a_f16(const half_t* base, int stride) {
  const int lane = threadIdx.x & 31;
  const int l = lane & 15;
  const int ko = (lane >> 4) ? 8 : 0;
  const half_t* q = base + l * stride + ko;
  H16 u;
  u.h[0] = *(const v8h*)(q);
  u.h[1] = *(const v8h*)(q + 16);
  return u.v;
}

// B fragment 32x16 (KxN), supplied as B^T row-major: [16 rows(N) x stride(K)].
// lanes 0-15 hold col n, K=0..7 & 16..23; lanes 16-31 hold col n, K=8..15 & 24..31
static __device__ __forceinline__ v16h load_bt_f16(const half_t* base, int stride) {
  const int lane = threadIdx.x & 31;
  const int l = lane & 15;
  const int ko = (lane >> 4) ? 8 : 0;
  const half_t* q = base + l * stride + ko;
  H16 u;
  u.h[0] = *(const v8h*)(q);
  u.h[1] = *(const v8h*)(q + 16);
  return u.v;
}

__global__ void cvt_f32_to_f16(const float* __restrict__ s, half_t* __restrict__ d, int n) {
  int i = (blockIdx.x * blockDim.x + threadIdx.x) * 4;
  if (i < n) {
    v4f f = *(const v4f*)(s + i);
    v4h h;
#pragma unroll
    for (int j = 0; j < 4; j++) h[j] = (_Float16)f[j];
    *(v4h*)(d + i) = h;
  }
}

// ---------------------------------------------------------------------------
// GEMM: D[m,n] = sum_k A[m,k] * W[n,k] + bias[n]
// Block tile 128x128, K-step 32, 8 waves, wave computes 32(M) x 64(N).
// Double-buffered global->LDS staging via CDNA5 async-to-LDS DMA (ASYNCcnt):
// tile i+1 streams into LDS while tile i runs through the WMMAs.
// MODE 0: f16 head-split Q/K [B,H,S,DK]; MODE 1: f16 V^T [B,H,DK,S];
// MODE 2: f32 plain [M, D_MODEL].
// ---------------------------------------------------------------------------
#define TM 128
#define TN 128
#define TK 32
#define LDSS 40   // padded LDS row stride in halfs (80B -> conflict-free b128 reads)

static __device__ __forceinline__ void async_ld16(half_t* lds_dst, const half_t* gsrc) {
  unsigned lds_off = (unsigned)(uintptr_t)lds_dst;   // low 32b of LDS aperture = offset
  asm volatile("global_load_async_to_lds_b128 %0, %1, off"
               :: "v"(lds_off), "v"(gsrc) : "memory");
}

template<int MODE>
__global__ __launch_bounds__(256)
void gemm_kernel(const half_t* __restrict__ A, const half_t* __restrict__ Bw,
                 const float* __restrict__ bias, void* __restrict__ outp) {
  __shared__ half_t As[2][TM * LDSS];
  __shared__ half_t Bs[2][TN * LDSS];
  const int mBase = blockIdx.x * TM;
  const int nBase = blockIdx.y * TN;
  const int wave  = threadIdx.x >> 5;
  const int wm    = (wave >> 1) * 32;   // 4 waves along M
  const int wn    = (wave & 1) * 64;    // 2 waves along N
  const int lane  = threadIdx.x & 31;
  const int l     = lane & 15;
  const int hi    = lane >> 4;

  const v8f vz = {0.f,0.f,0.f,0.f,0.f,0.f,0.f,0.f};
  v8f acc[2][4];
#pragma unroll
  for (int i = 0; i < 2; i++)
#pragma unroll
    for (int j = 0; j < 4; j++) acc[i][j] = vz;

  // per-thread staging assignment: 512 x 16B chunks per tile, 256 threads x 2
  const int c0  = threadIdx.x;
  const int r0  = c0 >> 2,        kc0 = (c0 & 3) * 8;
  const int c1  = threadIdx.x + 256;
  const int r1  = c1 >> 2,        kc1 = (c1 & 3) * 8;
  const half_t* gA0 = A  + (size_t)(mBase + r0) * D_MODEL + kc0;
  const half_t* gB0 = Bw + (size_t)(nBase + r0) * D_MODEL + kc0;
  const half_t* gA1 = A  + (size_t)(mBase + r1) * D_MODEL + kc1;
  const half_t* gB1 = Bw + (size_t)(nBase + r1) * D_MODEL + kc1;

  // prologue: stream first K-tile into buffer 0
  async_ld16(As[0] + r0 * LDSS + kc0, gA0);
  async_ld16(Bs[0] + r0 * LDSS + kc0, gB0);
  async_ld16(As[0] + r1 * LDSS + kc1, gA1);
  async_ld16(Bs[0] + r1 * LDSS + kc1, gB1);

  const int NSTEP = D_MODEL / TK;
  for (int it = 0; it < NSTEP; it++) {
    const int cur = it & 1;
    const int nxt = cur ^ 1;
    // own async loads for current buffer complete:
    asm volatile("s_wait_asynccnt 0" ::: "memory");
    // all waves' loads visible; all waves done consuming the other buffer
    // (their fragment DS reads were consumed by WMMAs before reaching here):
    __syncthreads();
    if (it + 1 < NSTEP) {
      const int kn = (it + 1) * TK;
      async_ld16(As[nxt] + r0 * LDSS + kc0, gA0 + kn);
      async_ld16(Bs[nxt] + r0 * LDSS + kc0, gB0 + kn);
      async_ld16(As[nxt] + r1 * LDSS + kc1, gA1 + kn);
      async_ld16(Bs[nxt] + r1 * LDSS + kc1, gB1 + kn);
    }

    v16h a0 = load_a_f16(As[cur] + (wm +  0) * LDSS, LDSS);
    v16h a1 = load_a_f16(As[cur] + (wm + 16) * LDSS, LDSS);
    v16h b0 = load_bt_f16(Bs[cur] + (wn +  0) * LDSS, LDSS);
    v16h b1 = load_bt_f16(Bs[cur] + (wn + 16) * LDSS, LDSS);
    v16h b2 = load_bt_f16(Bs[cur] + (wn + 32) * LDSS, LDSS);
    v16h b3 = load_bt_f16(Bs[cur] + (wn + 48) * LDSS, LDSS);
    acc[0][0] = wmma16x16x32(a0, b0, acc[0][0]);
    acc[1][0] = wmma16x16x32(a1, b0, acc[1][0]);
    acc[0][1] = wmma16x16x32(a0, b1, acc[0][1]);
    acc[1][1] = wmma16x16x32(a1, b1, acc[1][1]);
    acc[0][2] = wmma16x16x32(a0, b2, acc[0][2]);
    acc[1][2] = wmma16x16x32(a1, b2, acc[1][2]);
    acc[0][3] = wmma16x16x32(a0, b3, acc[0][3]);
    acc[1][3] = wmma16x16x32(a1, b3, acc[1][3]);
  }

  // Epilogue. C layout: VGPR j -> row (tile + j + hi*8), col (tile + l).
#pragma unroll
  for (int mt = 0; mt < 2; mt++) {
#pragma unroll
    for (int nt = 0; nt < 4; nt++) {
      const int n  = nBase + wn + nt * 16 + l;
      const float bv = bias[n];
      if (MODE == 1) {
        // V^T[b,h,d,s]: 8 accumulator rows are 8 consecutive tokens -> one b128
        const int h = n >> 6, d = n & 63;
        const int tok0 = mBase + wm + mt * 16 + hi * 8;
        const int b  = tok0 >> 11, s0 = tok0 & (SEQ - 1);
        v8h pk;
#pragma unroll
        for (int j = 0; j < 8; j++) pk[j] = (_Float16)(acc[mt][nt][j] + bv);
        half_t* dst = (half_t*)outp +
                      ((size_t)((b * NHEAD + h) * DK + d)) * SEQ + s0;
        *(v8h*)dst = pk;
      } else {
#pragma unroll
        for (int j = 0; j < 8; j++) {
          const int row = mBase + wm + mt * 16 + hi * 8 + j;
          const float val = acc[mt][nt][j] + bv;
          if (MODE == 0) {
            const int h = n >> 6, d = n & 63;
            const int b = row >> 11, s = row & (SEQ - 1);
            ((half_t*)outp)[((size_t)((b * NHEAD + h) * SEQ + s)) * DK + d] =
                (half_t)val;
          } else {
            ((float*)outp)[(size_t)row * D_MODEL + n] = val;
          }
        }
      }
    }
  }
}

// ---------------------------------------------------------------------------
// Banded flash attention. One wave per 16-query tile per (b,h).
// Q,K: [B,H,S,64] f16 ; Vt: [B,H,64,S] f16 ; out: attn [NTOK, D_MODEL] f16
// ---------------------------------------------------------------------------
__global__ __launch_bounds__(128)
void attn_kernel(const half_t* __restrict__ Q, const half_t* __restrict__ K,
                 const half_t* __restrict__ Vt, half_t* __restrict__ Ao) {
  __shared__ float p_lds[4][16 * 36];   // per-wave 16x32 P tile, padded stride 36
  const int wlocal = threadIdx.x >> 5;
  const int gid = blockIdx.x * 4 + wlocal;     // 0 .. 4095
  const int qt  = gid & (SEQ / 16 - 1);
  const int bh  = gid >> 7;                    // b*16 + h, 0..31
  const int q0  = qt * 16;
  const int lane = threadIdx.x & 31;
  const int l   = lane & 15;
  const int hi  = lane >> 4;

  const half_t* Qb = Q  + ((size_t)bh * SEQ + q0) * DK;
  const half_t* Kb = K  + (size_t)bh * SEQ * DK;
  const half_t* Vb = Vt + (size_t)bh * DK * SEQ;
  float* P = p_lds[wlocal];

  const v16h qf0 = load_a_f16(Qb,      DK);    // d = 0..31
  const v16h qf1 = load_a_f16(Qb + 32, DK);    // d = 32..63

  const v8f vz = {0.f,0.f,0.f,0.f,0.f,0.f,0.f,0.f};
  v8f o[4];
#pragma unroll
  for (int i = 0; i < 4; i++) o[i] = vz;
  float m_s[8], l_s[8];
#pragma unroll
  for (int r = 0; r < 8; r++) { m_s[r] = -1e9f; l_s[r] = 0.f; }

  int kstart = q0 - (BAND - 1); if (kstart < 0) kstart = 0;
  kstart &= ~31;
  int kend = q0 + 15 + BAND; if (kend > SEQ) kend = SEQ;

  for (int kc = kstart; kc < kend; kc += 32) {
    // ---- S = Q K^T : 16 queries x 32 keys; issue all loads, then 4 WMMAs ----
    const half_t* kr = Kb + (size_t)kc * DK;
    v16h kb0 = load_bt_f16(kr,            DK);
    v16h kb1 = load_bt_f16(kr + 32,       DK);
    v16h kb2 = load_bt_f16(kr + 16 * DK,      DK);
    v16h kb3 = load_bt_f16(kr + 16 * DK + 32, DK);
    v8f s0 = vz, s1 = vz;
    s0 = wmma16x16x32(qf0, kb0, s0);
    s1 = wmma16x16x32(qf0, kb2, s1);
    s0 = wmma16x16x32(qf1, kb1, s0);
    s1 = wmma16x16x32(qf1, kb3, s1);

    const float scale = 0.125f;           // 1/sqrt(DK)
    const int rowb = q0 + hi * 8;
    const int key0 = kc + l;
    float alpha[8];
#pragma unroll
    for (int r = 0; r < 8; r++) {
      const int rr = rowb + r;
      const int d0 = rr - key0;
      const int d1 = rr - (key0 + 16);
      float v0 = (d0 < BAND && d0 > -BAND) ? s0[r] * scale : -1e9f;
      float v1 = (d1 < BAND && d1 > -BAND) ? s1[r] * scale : -1e9f;
      float mx = rowmax16(fmaxf(v0, v1));           // DPP16 butterfly, pure VALU
      const float nm = fmaxf(m_s[r], mx);
      const float al = __expf(m_s[r] - nm);
      m_s[r] = nm;
      v0 = __expf(v0 - nm);
      v1 = __expf(v1 - nm);
      const float rs = rowsum16(v0 + v1);
      l_s[r] = l_s[r] * al + rs;
      alpha[r] = al;
      // spill P (C layout) to LDS for layout change to A-fragment
      P[(r + hi * 8) * 36 + l]      = v0;
      P[(r + hi * 8) * 36 + 16 + l] = v1;
    }
#pragma unroll
    for (int nt = 0; nt < 4; nt++)
#pragma unroll
      for (int r = 0; r < 8; r++) o[nt][r] *= alpha[r];

    // same-wave DS ordering + compiler barrier before re-reading P transposed
    asm volatile("s_wait_dscnt 0" ::: "memory");

    // A fragment (16x32) of P, f32 LDS -> f16 regs
    v16h pf;
    {
      const float* prow = P + l * 36 + (hi ? 8 : 0);
      v4f f0 = *(const v4f*)(prow + 0);
      v4f f1 = *(const v4f*)(prow + 4);
      v4f f2 = *(const v4f*)(prow + 16);
      v4f f3 = *(const v4f*)(prow + 20);
#pragma unroll
      for (int i = 0; i < 4; i++) {
        pf[i]      = (_Float16)f0[i];
        pf[4 + i]  = (_Float16)f1[i];
        pf[8 + i]  = (_Float16)f2[i];
        pf[12 + i] = (_Float16)f3[i];
      }
    }

    // ---- O += P * V : issue all 4 V fragments, then 4 WMMAs ----
    v16h vb0 = load_bt_f16(Vb + (size_t)( 0) * SEQ + kc, SEQ);
    v16h vb1 = load_bt_f16(Vb + (size_t)(16) * SEQ + kc, SEQ);
    v16h vb2 = load_bt_f16(Vb + (size_t)(32) * SEQ + kc, SEQ);
    v16h vb3 = load_bt_f16(Vb + (size_t)(48) * SEQ + kc, SEQ);
    o[0] = wmma16x16x32(pf, vb0, o[0]);
    o[1] = wmma16x16x32(pf, vb1, o[1]);
    o[2] = wmma16x16x32(pf, vb2, o[2]);
    o[3] = wmma16x16x32(pf, vb3, o[3]);
  }

  // finalize: O /= l ; store to attn[token, h*64+d] (f16, input of Wo GEMM)
  const int h = bh & 15;
  const int b = bh >> 4;
#pragma unroll
  for (int r = 0; r < 8; r++) {
    const float inv = 1.0f / l_s[r];
    const int tok = q0 + hi * 8 + r;
    half_t* dst = Ao + ((size_t)(b * SEQ + tok)) * D_MODEL + h * DK;
#pragma unroll
    for (int nt = 0; nt < 4; nt++)
      dst[nt * 16 + l] = (half_t)(o[nt][r] * inv);
  }
}

// ---------------------------------------------------------------------------
extern "C" void kernel_launch(void* const* d_in, const int* in_sizes, int n_in,
                              void* d_out, int out_size, void* d_ws, size_t ws_size,
                              hipStream_t stream) {
  const float* x  = (const float*)d_in[0];
  const float* Wq = (const float*)d_in[1];
  const float* bq = (const float*)d_in[2];
  const float* Wk = (const float*)d_in[3];
  const float* bk = (const float*)d_in[4];
  const float* Wv = (const float*)d_in[5];
  const float* bv = (const float*)d_in[6];
  const float* Wo = (const float*)d_in[7];
  const float* bo = (const float*)d_in[8];

  // Workspace (f16): Xh | Wqh | Wkh | Wvh | Woh | Qh | Kh | Vt | attn  = 48 MB
  const size_t NX = (size_t)NTOK * D_MODEL;
  const size_t NW = (size_t)D_MODEL * D_MODEL;
  half_t* ws  = (half_t*)d_ws;
  half_t* Xh  = ws;  ws += NX;
  half_t* Wqh = ws;  ws += NW;
  half_t* Wkh = ws;  ws += NW;
  half_t* Wvh = ws;  ws += NW;
  half_t* Woh = ws;  ws += NW;
  half_t* Qh  = ws;  ws += NX;
  half_t* Kh  = ws;  ws += NX;
  half_t* Vth = ws;  ws += NX;
  half_t* Ah  = ws;  ws += NX;

  cvt_f32_to_f16<<<(unsigned)((NX / 4 + 255) / 256), 256, 0, stream>>>(x,  Xh,  (int)NX);
  cvt_f32_to_f16<<<(unsigned)((NW / 4 + 255) / 256), 256, 0, stream>>>(Wq, Wqh, (int)NW);
  cvt_f32_to_f16<<<(unsigned)((NW / 4 + 255) / 256), 256, 0, stream>>>(Wk, Wkh, (int)NW);
  cvt_f32_to_f16<<<(unsigned)((NW / 4 + 255) / 256), 256, 0, stream>>>(Wv, Wvh, (int)NW);
  cvt_f32_to_f16<<<(unsigned)((NW / 4 + 255) / 256), 256, 0, stream>>>(Wo, Woh, (int)NW);

  dim3 gg(NTOK / TM, D_MODEL / TN);   // 32 x 8
  gemm_kernel<0><<<gg, 256, 0, stream>>>(Xh, Wqh, bq, (void*)Qh);
  gemm_kernel<0><<<gg, 256, 0, stream>>>(Xh, Wkh, bk, (void*)Kh);
  gemm_kernel<1><<<gg, 256, 0, stream>>>(Xh, Wvh, bv, (void*)Vth);

  attn_kernel<<<(BATCH * NHEAD * (SEQ / 16)) / 4, 128, 0, stream>>>(Qh, Kh, Vth, Ah);

  gemm_kernel<2><<<gg, 256, 0, stream>>>(Ah, Woh, bo, d_out);
}
